// LearnableFrontEnd_77824807403755
// MI455X (gfx1250) — matrix operand order repaired and checked by reading, not compiled, not run on hardware
//
#include <hip/hip_runtime.h>
#include <math.h>

typedef __attribute__((ext_vector_type(16))) _Float16 v16h;
typedef __attribute__((ext_vector_type(4)))  _Float16 h4;
typedef __attribute__((ext_vector_type(8)))  float    v8f;

#define SR_F      22050.0f
#define WIN       551
#define STRIDE    220
#define NF        40
#define NTILE_N   3            // 48 padded filters / 16
#define KCHUNKS   18           // ceil(551/32) -> K padded to 576
#define B_BATCH   32
#define N_SAMP    661500
#define TOTAL     (B_BATCH * N_SAMP)     // 21,168,000
#define PAD_H     1024                    // zeroed tail halves (covers K/time pad)
#define TOTAL_P   (TOTAL + PAD_H)
#define T_CONV    3005
#define T_TILES   188          // ceil(3005/16)
#define T_POOL    1502
#define TWO_PI    6.283185307179586f

// ---------------------------------------------------------------------------
// Kernel 0: split waveform into f16 hi/lo pair (error-compensated source for
// the WMMA conv).  Pure streaming pass; also zeroes the tail pad.
// ---------------------------------------------------------------------------
__global__ __launch_bounds__(256) void split_kernel(
    const float* __restrict__ wav,
    _Float16* __restrict__ hi, _Float16* __restrict__ lo) {
  size_t i4 = ((size_t)blockIdx.x * 256 + threadIdx.x) * 4;
  if (i4 >= (size_t)TOTAL_P) return;
  h4 hv, lv;
  if (i4 < (size_t)TOTAL) {
    float4 v = *(const float4*)(wav + i4);
    float x[4] = {v.x, v.y, v.z, v.w};
#pragma unroll
    for (int j = 0; j < 4; ++j) {
      _Float16 h = (_Float16)x[j];
      hv[j] = h;
      lv[j] = (_Float16)(x[j] - (float)h);
    }
  } else {
    hv = (h4)(_Float16)0.0f;
    lv = (h4)(_Float16)0.0f;
  }
  *(h4*)(hi + i4) = hv;
  *(h4*)(lo + i4) = lv;
}

// ---------------------------------------------------------------------------
// Kernel 1: evaluate Gabor filters, pack hi/lo f16 in WMMA B-fragment layout
// (32x16 f16, wave32): lane holds column n=lane&15, element i holds
// K = i + 8*(i>=8) + 8*(lane>=16).  Fragment index = (nt*18+c)*32+lane.
// ---------------------------------------------------------------------------
__global__ __launch_bounds__(32) void gabor_prep_kernel(
    const float* __restrict__ cf, const float* __restrict__ bw,
    _Float16* __restrict__ Bhi, _Float16* __restrict__ Blo) {
  int blk  = blockIdx.x;                 // 0..53 = nt*18 + c
  int lane = threadIdx.x;
  int nt   = blk / KCHUNKS;
  int c    = blk % KCHUNKS;
  int n    = nt * 16 + (lane & 15);
  int kbase = c * 32 + ((lane >= 16) ? 8 : 0);

  float cfn = (n < NF) ? cf[n] : 0.0f;
  float bwn = (n < NF) ? bw[n] : 1.0f;
  float omega     = TWO_PI * cfn;
  float inv_sigma = TWO_PI * bwn;        // sigma = 1/(2*pi*bw)

  v16h hi, lo;
#pragma unroll
  for (int i = 0; i < 16; ++i) {
    int k = kbase + i + ((i >= 8) ? 8 : 0);
    float w = 0.0f;
    if (n < NF && k < WIN) {             // zero pad: kills OOB A columns too
      float t = (float)k * (1.0f / SR_F);
      float z = t * inv_sigma;
      w = expf(-0.5f * z * z) * cosf(omega * t);
    }
    _Float16 h = (_Float16)w;
    hi[i] = h;
    lo[i] = (_Float16)(w - (float)h);
  }
  size_t off = ((size_t)blk * 32 + lane) * 16;
  *(v16h*)(Bhi + off) = hi;
  *(v16h*)(Blo + off) = lo;
}

// ---------------------------------------------------------------------------
// Kernel 2: strided conv as WMMA GEMM, no LDS.  Block = 256 threads = 8 waves,
// one time-tile per wave.  A hi/lo fragments are 4x b64 loads each, straight
// from the pre-split f16 arrays (immediate offsets after full unroll); each
// wave drives 3 accumulators (48 padded filters) -> 9 WMMAs per K-chunk.
// Epilogue squares + AvgPool(2) and writes chain-major E[b*40+n][tp].
// ---------------------------------------------------------------------------
__global__ __launch_bounds__(256) void gabor_conv_wmma_kernel(
    const _Float16* __restrict__ wavHi, const _Float16* __restrict__ wavLo,
    const _Float16* __restrict__ Bhi,   const _Float16* __restrict__ Blo,
    float* __restrict__ E) {
  int wave  = threadIdx.x >> 5;
  int lane  = threadIdx.x & 31;
  int ttile = blockIdx.x * 8 + wave;     // 0..191
  int b     = blockIdx.y;                // 0..31
  if (ttile >= T_TILES) return;          // wave-uniform

  int m     = lane & 15;                 // time row within tile (A layout)
  int koffA = (lane >= 16) ? 8 : 0;      // A-layout K interleave
  size_t rowbase = (size_t)b * N_SAMP + (size_t)(ttile * 16 + m) * STRIDE + koffA;
  const _Float16* pah = wavHi + rowbase;
  const _Float16* pal = wavLo + rowbase;
  const v16h* BH = (const v16h*)Bhi + lane;   // + fragment index * 32 below
  const v16h* BL = (const v16h*)Blo + lane;

  v8f acc0 = {}, acc1 = {}, acc2 = {};
#pragma unroll
  for (int c = 0; c < KCHUNKS; ++c) {
    int kb = c * 32;
    // A fragment: elems 0..7 = halves [kb, kb+8), elems 8..15 = [kb+16, kb+24)
    v16h ahi, alo;
    {
      h4 p0 = *(const h4*)(pah + kb);
      h4 p1 = *(const h4*)(pah + kb + 4);
      h4 p2 = *(const h4*)(pah + kb + 16);
      h4 p3 = *(const h4*)(pah + kb + 20);
      h4 q0 = *(const h4*)(pal + kb);
      h4 q1 = *(const h4*)(pal + kb + 4);
      h4 q2 = *(const h4*)(pal + kb + 16);
      h4 q3 = *(const h4*)(pal + kb + 20);
#pragma unroll
      for (int j = 0; j < 4; ++j) {
        ahi[j]      = p0[j]; ahi[4 + j]  = p1[j];
        ahi[8 + j]  = p2[j]; ahi[12 + j] = p3[j];
        alo[j]      = q0[j]; alo[4 + j]  = q1[j];
        alo[8 + j]  = q2[j]; alo[12 + j] = q3[j];
      }
    }
    v16h bh0 = BH[(0 * KCHUNKS + c) * 32];
    v16h bh1 = BH[(1 * KCHUNKS + c) * 32];
    v16h bh2 = BH[(2 * KCHUNKS + c) * 32];
    v16h bl0 = BL[(0 * KCHUNKS + c) * 32];
    v16h bl1 = BL[(1 * KCHUNKS + c) * 32];
    v16h bl2 = BL[(2 * KCHUNKS + c) * 32];
    acc0 = __builtin_amdgcn_wmma_f32_16x16x32_f16(false, ahi, false, bh0,
                                                  (short)0, acc0, false, false);
    acc1 = __builtin_amdgcn_wmma_f32_16x16x32_f16(false, ahi, false, bh1,
                                                  (short)0, acc1, false, false);
    acc2 = __builtin_amdgcn_wmma_f32_16x16x32_f16(false, ahi, false, bh2,
                                                  (short)0, acc2, false, false);
    acc0 = __builtin_amdgcn_wmma_f32_16x16x32_f16(false, ahi, false, bl0,
                                                  (short)0, acc0, false, false);
    acc1 = __builtin_amdgcn_wmma_f32_16x16x32_f16(false, ahi, false, bl1,
                                                  (short)0, acc1, false, false);
    acc2 = __builtin_amdgcn_wmma_f32_16x16x32_f16(false, ahi, false, bl2,
                                                  (short)0, acc2, false, false);
    acc0 = __builtin_amdgcn_wmma_f32_16x16x32_f16(false, alo, false, bh0,
                                                  (short)0, acc0, false, false);
    acc1 = __builtin_amdgcn_wmma_f32_16x16x32_f16(false, alo, false, bh1,
                                                  (short)0, acc1, false, false);
    acc2 = __builtin_amdgcn_wmma_f32_16x16x32_f16(false, alo, false, bh2,
                                                  (short)0, acc2, false, false);
  }

  // D layout: col n = lane&15, row (time) = v (+8 for lanes>=16)
  int tpbase = ttile * 8 + ((lane >= 16) ? 4 : 0);
  v8f accs[3] = {acc0, acc1, acc2};
#pragma unroll
  for (int nt = 0; nt < NTILE_N; ++nt) {
    int n = nt * 16 + (lane & 15);
    if (n < NF) {
      float* Echain = E + (size_t)(b * NF + n) * T_POOL;
#pragma unroll
      for (int j = 0; j < 4; ++j) {
        float e0 = accs[nt][2 * j]     * accs[nt][2 * j];
        float e1 = accs[nt][2 * j + 1] * accs[nt][2 * j + 1];
        int tp = tpbase + j;
        if (tp < T_POOL) Echain[tp] = 0.5f * (e0 + e1);
      }
    }
  }
}

// ---------------------------------------------------------------------------
// Kernel 3: PCEN + log as a wave-parallel scan.  One wave per (b,c) chain
// (1280 waves).  Constant-coefficient EMA closed form per 32-step chunk:
//   s_j = alpha^j * (alpha*carry + 0.02 * prefix_j(alpha^-i * x_i))
// prefix via 5 shfl_up steps; loads/stores fully coalesced (chain-major).
// ---------------------------------------------------------------------------
__global__ __launch_bounds__(128) void pcen_scan_kernel(
    const float* __restrict__ E, float* __restrict__ out) {
  int gtid  = blockIdx.x * 128 + threadIdx.x;
  int chain = gtid >> 5;                  // 0..1279 = b*40 + c
  int lane  = gtid & 31;
  const float* x = E + (size_t)chain * T_POOL;
  float*       o = out + (size_t)chain * T_POOL;

  const float alpha = 0.98f, beta = 0.02f, eps = 1e-6f;
  const float sqrt2 = 1.4142135623730951f;
  float lnA  = logf(alpha);
  float aj   = expf((float)lane * lnA);   // alpha^lane
  float iaj  = expf(-(float)lane * lnA);  // alpha^-lane
  float carry = x[0];                     // s_{-1}=x0  =>  s_0 = x_0 exactly

  for (int c0 = 0; c0 < T_POOL; c0 += 32) {
    int t = c0 + lane;
    float xv = (t < T_POOL) ? x[t] : 0.0f;
    float z = xv * iaj;
#pragma unroll
    for (int d = 1; d < 32; d <<= 1) {
      float up = __shfl_up(z, d, 32);
      if (lane >= d) z += up;
    }
    float s = aj * (alpha * carry + beta * z);
    carry = __shfl(s, 31, 32);
    float invp = expf(-alpha * logf(s + eps));       // (s+eps)^-alpha
    float p = sqrtf(xv * invp + 2.0f) - sqrt2;
    if (t < T_POOL) o[t] = logf(p + 1e-8f);
  }
}

// ---------------------------------------------------------------------------
extern "C" void kernel_launch(void* const* d_in, const int* in_sizes, int n_in,
                              void* d_out, int out_size, void* d_ws, size_t ws_size,
                              hipStream_t stream) {
  const float* wav = (const float*)d_in[0];   // [32, 661500]
  const float* cf  = (const float*)d_in[1];   // [40]
  const float* bw  = (const float*)d_in[2];   // [40]
  float* out = (float*)d_out;                 // [32, 40, 1502]

  char* ws = (char*)d_ws;
  const size_t wav16_bytes = (size_t)TOTAL_P * sizeof(_Float16);   // 42,338,048
  const size_t frag_bytes  = (size_t)NTILE_N * KCHUNKS * 32 * 16 * sizeof(_Float16); // 55,296
  _Float16* wavHi = (_Float16*)ws;
  _Float16* wavLo = (_Float16*)(ws + wav16_bytes);
  _Float16* Bhi   = (_Float16*)(ws + 2 * wav16_bytes);
  _Float16* Blo   = (_Float16*)(ws + 2 * wav16_bytes + frag_bytes);
  float*    E     = (float*)(ws + 2 * wav16_bytes + 2 * frag_bytes); // ~7.7 MB

  split_kernel<<<(TOTAL_P / 4 + 255) / 256, 256, 0, stream>>>(wav, wavHi, wavLo);
  gabor_prep_kernel<<<NTILE_N * KCHUNKS, 32, 0, stream>>>(cf, bw, Bhi, Blo);
  gabor_conv_wmma_kernel<<<dim3(24, B_BATCH), 256, 0, stream>>>(wavHi, wavLo, Bhi, Blo, E);
  pcen_scan_kernel<<<(B_BATCH * NF * 32) / 128, 128, 0, stream>>>(E, out);
}
